// EZPairEmbedding_37099927502920
// MI455X (gfx1250) — matrix operand order repaired and checked by reading, not compiled, not run on hardware
//
#include <hip/hip_runtime.h>

typedef __attribute__((ext_vector_type(4)))  float  v4f;
typedef __attribute__((ext_vector_type(8)))  float  v8f;
typedef __attribute__((ext_vector_type(4)))  __bf16 v4bf;
typedef __attribute__((ext_vector_type(8)))  __bf16 v8bf;
typedef __attribute__((ext_vector_type(16))) __bf16 v16bf;

#define NWG        512
#define TPB        256
#define SMEM_BYTES 131072   // 64KB W1 frags + 32KB W2 frags + 8*4KB per-wave h-buffers

// Build a 16-bit A-matrix fragment (16x32, this lane's 16 bf16 values) from two
// 8-float contiguous chunks (c0 -> elements 0..7, c1 -> elements 8..15).
__device__ __forceinline__ v16bf make_a_from_f32(const float* c0, const float* c1) {
  v4f f0 = *(const v4f*)(c0);
  v4f f1 = *(const v4f*)(c0 + 4);
  v4f f2 = *(const v4f*)(c1);
  v4f f3 = *(const v4f*)(c1 + 4);
  v4bf g0 = __builtin_convertvector(f0, v4bf);
  v4bf g1 = __builtin_convertvector(f1, v4bf);
  v4bf g2 = __builtin_convertvector(f2, v4bf);
  v4bf g3 = __builtin_convertvector(f3, v4bf);
  v8bf lo = __builtin_shufflevector(g0, g1, 0,1,2,3,4,5,6,7);
  v8bf hi = __builtin_shufflevector(g2, g3, 0,1,2,3,4,5,6,7);
  return __builtin_shufflevector(lo, hi, 0,1,2,3,4,5,6,7,8,9,10,11,12,13,14,15);
}

__device__ __forceinline__ v16bf make_a_from_lds(const __bf16* c0) {
  v8bf lo = *(const v8bf*)(c0);
  v8bf hi = *(const v8bf*)(c0 + 8);
  return __builtin_shufflevector(lo, hi, 0,1,2,3,4,5,6,7,8,9,10,11,12,13,14,15);
}

__global__ __launch_bounds__(TPB) void pair_embed_wmma(
    const int* __restrict__ aa, const int* __restrict__ res_nb,
    const unsigned char* __restrict__ mask_atoms,
    const float* __restrict__ aa_pair_table, const float* __restrict__ relpos_table,
    const float* __restrict__ W1, const float* __restrict__ b1,
    const float* __restrict__ W2, const float* __restrict__ b2,
    float* __restrict__ out)
{
  extern __shared__ __align__(32) char smem[];
  __bf16* w1f = (__bf16*)smem;                 // 32768 bf16 = 64 fragments (s=0..7, t=0..7)
  __bf16* w2f = (__bf16*)(smem + 65536);       // 16384 bf16 = 32 fragments (s=0..3, t=0..7)

  const int tid = threadIdx.x;

  // ---- Pre-swizzle weights into B-fragment layout (bf16) in LDS ----
  // Fragment (s,t) covers K = 32s..32s+31, N = 16t..16t+15.
  // Within a fragment: lane = 16*(kk/16) + nn ; element index kr = kk%16 (2 per VGPR).
  for (int e = tid; e < 32768; e += TPB) {     // W1: (k,n) row-major, k in [0,256)
    int k = e >> 7, n = e & 127;
    int s = k >> 5, kk = k & 31, t = n >> 4, nn = n & 15;
    int lane = ((kk >> 4) << 4) + nn;
    w1f[((s * 8 + t) << 9) + (lane << 4) + (kk & 15)] = (__bf16)W1[e];
  }
  for (int e = tid; e < 16384; e += TPB) {     // W2: (k,n) row-major, k in [0,128)
    int k = e >> 7, n = e & 127;
    int s = k >> 5, kk = k & 31, t = n >> 4, nn = n & 15;
    int lane = ((kk >> 4) << 4) + nn;
    w2f[((s * 8 + t) << 9) + (lane << 4) + (kk & 15)] = (__bf16)W2[e];
  }
  __syncthreads();

  const int lane   = tid & 31;
  const int waveId = tid >> 5;
  const int row    = lane & 15;   // A-matrix row this lane feeds
  const int half   = lane >> 4;   // which K-half this lane holds
  __bf16* hbuf = (__bf16*)(smem + 98304) + waveId * 2048;  // 16x128 bf16, row-major

  float b1v[8], b2v[8];
#pragma unroll
  for (int t = 0; t < 8; ++t) {
    b1v[t] = b1[t * 16 + row];
    b2v[t] = b2[t * 16 + row];
  }

  const int nTiles = (8 * 256 * 256) / 16;     // 32768 = NWG*8 waves * 8 iters (uniform)
  for (int tileId = blockIdx.x * 8 + waveId; tileId < nTiles; tileId += NWG * 8) {
    // ---- per-row gather setup (lane pair rows: p = tile*16 + row) ----
    int p = tileId * 16 + row;
    int n = p >> 16, i = (p >> 8) & 255, j = p & 255;
    int gi = (n << 8) + i, gj = (n << 8) + j;
    int aai = aa[gi], aaj = aa[gj];
    const float* aaBase = aa_pair_table + (size_t)(aai * 22 + aaj) * 128;
    int rp = res_nb[gi] - res_nb[gj];
    rp = (rp < -32) ? -32 : (rp > 32 ? 32 : rp);
    const float* rpBase = relpos_table + (size_t)(rp + 32) * 128;
    bool mB = (mask_atoms[gi * 15 + 1] != 0) && (mask_atoms[gj * 15 + 1] != 0);
    unsigned mbits = __builtin_amdgcn_ballot_w32(mB);   // bit r == mask of row r (0..15)

    // ---- layer 1: h = relu(X @ W1 + b1), X gathered on the fly ----
    v8f acc[8] = {};
#pragma unroll
    for (int s = 0; s < 8; ++s) {
      const float* src = (s < 4) ? (aaBase + 32 * s) : (rpBase + 32 * s - 128);
      const float* c0 = src + 16 * half;       // K = k0 + 16*half .. +7
      const float* c1 = c0 + 8;                // K = k0 + 16*half + 8 .. +7
      v16bf A = make_a_from_f32(c0, c1);
      const __bf16* fb = w1f + (s << 12) + (lane << 4);
#pragma unroll
      for (int t = 0; t < 8; ++t) {
        v16bf B = *(const v16bf*)(fb + (t << 9));
        acc[t] = __builtin_amdgcn_wmma_f32_16x16x32_bf16(
            false, A, false, B, (short)0, acc[t], false, false);
      }
    }

    // ---- relu + bias, spill h tile to LDS (bf16, row-major) for C->A transpose ----
#pragma unroll
    for (int t = 0; t < 8; ++t) {
#pragma unroll
      for (int v = 0; v < 8; ++v) {
        float x = acc[t][v] + b1v[t];
        x = x > 0.f ? x : 0.f;
        hbuf[(v + 8 * half) * 128 + t * 16 + row] = (__bf16)x;
      }
    }
    asm volatile("s_wait_dscnt 0" ::: "memory");

    // ---- layer 2: out = h @ W2 + b2 ----
    v8f acc2[8] = {};
#pragma unroll
    for (int s = 0; s < 4; ++s) {
      const __bf16* h0 = hbuf + row * 128 + s * 32 + 16 * half;
      v16bf A = make_a_from_lds(h0);
      const __bf16* fb = w2f + (s << 12) + (lane << 4);
#pragma unroll
      for (int t = 0; t < 8; ++t) {
        v16bf B = *(const v16bf*)(fb + (t << 9));
        acc2[t] = __builtin_amdgcn_wmma_f32_16x16x32_bf16(
            false, A, false, B, (short)0, acc2[t], false, false);
      }
    }

    // ---- bias, mask, store ----
    float* op = out + (size_t)tileId * 16 * 128;
#pragma unroll
    for (int t = 0; t < 8; ++t) {
      int d = t * 16 + row;                    // C layout: N = lane%16 (+16t)
#pragma unroll
      for (int v = 0; v < 8; ++v) {
        int M = v + 8 * half;                  // C layout: M = v + 8*(lane/16)
        float val = acc2[t][v] + b2v[t];
        float sc = (float)((mbits >> M) & 1u);
        op[M * 128 + d] = val * sc;
      }
    }
  }
}

extern "C" void kernel_launch(void* const* d_in, const int* in_sizes, int n_in,
                              void* d_out, int out_size, void* d_ws, size_t ws_size,
                              hipStream_t stream) {
  const int*           aa            = (const int*)d_in[0];
  const int*           res_nb        = (const int*)d_in[1];
  // d_in[2] = pos_atoms (unused by the reference computation)
  const unsigned char* mask_atoms    = (const unsigned char*)d_in[3];
  const float*         aa_pair_table = (const float*)d_in[4];
  const float*         relpos_table  = (const float*)d_in[5];
  const float*         W1            = (const float*)d_in[6];
  const float*         b1            = (const float*)d_in[7];
  const float*         W2            = (const float*)d_in[8];
  const float*         b2            = (const float*)d_in[9];
  float*               out           = (float*)d_out;

  pair_embed_wmma<<<NWG, TPB, SMEM_BYTES, stream>>>(
      aa, res_nb, mask_atoms, aa_pair_table, relpos_table, W1, b1, W2, b2, out);
}
